// GraphConvDist_31190052504134
// MI455X (gfx1250) — matrix operand order repaired and checked by reading, not compiled, not run on hardware
//
#include <hip/hip_runtime.h>

typedef __bf16 bf16_t;
typedef __attribute__((ext_vector_type(16))) __bf16 v16bf;
typedef __attribute__((ext_vector_type(8)))  float  v8f;

#define Hh   128
#define K1   384            // 3H
#define K2   256            // 2H
#define KT1  (K1 / 32)      // 12 k-steps for GEMM1 (edge)
#define KTH  (Hh / 32)      // 4  k-steps for H x H GEMMs
#define KT2  (K2 / 32)      // 8  k-steps for GEMM1 (node)
#define NT   (Hh / 16)      // 8  n-tiles of 16 columns
#define FRAG 512            // 32 lanes * 16 bf16 per (kt,nt) fragment
#define ROWS 2              // 16-row tiles per wave (A-reuse per B fragment)

// fragment element counts per swizzled weight
#define WE1_ELEMS (KT1 * NT * FRAG)   // 49152
#define WE2_ELEMS (KTH * NT * FRAG)   // 16384
#define WF1_ELEMS (KT2 * NT * FRAG)   // 32768
#define WF2_ELEMS (KTH * NT * FRAG)   // 16384

// ---------------------------------------------------------------------------
// Swizzle fp32 weights [K x 128] into bf16 B-fragment order:
// dst[((kt*NT + nt)*32 + lane)*16 + i] = W[k][n]
//   lane<16 : n = nt*16 + lane,      k = kt*32 + (i&7) + (i>>3)*16
//   lane>=16: n = nt*16 + lane - 16, k = kt*32 + 8 + (i&7) + (i>>3)*16
// ---------------------------------------------------------------------------
__device__ __forceinline__ void swizzle_one(const float* __restrict__ src,
                                            bf16_t* __restrict__ dst, int idx) {
  const int i    = idx & 15;
  const int lane = (idx >> 4) & 31;
  const int rest = idx >> 9;
  const int nt   = rest & 7;
  const int kt   = rest >> 3;
  const int n    = nt * 16 + (lane & 15);
  const int k    = kt * 32 + ((lane >> 4) ? 8 : 0) + (i & 7) + (i >> 3) * 16;
  dst[idx] = (bf16_t)src[k * Hh + n];
}

__global__ void prep_weights(const float* __restrict__ We1, const float* __restrict__ We2,
                             const float* __restrict__ Wf1, const float* __restrict__ Wf2,
                             bf16_t* __restrict__ We1s, bf16_t* __restrict__ We2s,
                             bf16_t* __restrict__ Wf1s, bf16_t* __restrict__ Wf2s) {
  int gid = blockIdx.x * blockDim.x + threadIdx.x;
  if (gid < WE1_ELEMS) { swizzle_one(We1, We1s, gid); return; }
  gid -= WE1_ELEMS;
  if (gid < WE2_ELEMS) { swizzle_one(We2, We2s, gid); return; }
  gid -= WE2_ELEMS;
  if (gid < WF1_ELEMS) { swizzle_one(Wf1, Wf1s, gid); return; }
  gid -= WF1_ELEMS;
  if (gid < WF2_ELEMS) { swizzle_one(Wf2, Wf2s, gid); }
}

__global__ void init_agg(unsigned int* __restrict__ p, int n) {
  int i = blockIdx.x * blockDim.x + threadIdx.x;
  if (i < n) p[i] = 0xFF800000u;  // -inf
}

// sign-aware float atomic max via integer atomics (works with -inf init)
__device__ __forceinline__ void atomic_fmax(float* addr, float v) {
  if (v >= 0.0f) atomicMax((int*)addr, __float_as_int(v));
  else           atomicMin((unsigned int*)addr, __float_as_uint(v));
}

// Build a 16x32 bf16 A-fragment from a row-major fp32 row.
// Lane layout (ISA 7.12.2): lanes 0-15 row m hold K = k0..k0+7, k0+16..k0+23;
// lanes 16-31 hold K = k0+8..k0+15, k0+24..k0+31  (k0 already includes +8 for hi).
__device__ __forceinline__ v16bf make_a_f32(const float* __restrict__ rowp,
                                            int k0, bool ok) {
  v16bf a;
  if (ok) {
    const float4 c00 = ((const float4*)(rowp + k0))[0];
    const float4 c01 = ((const float4*)(rowp + k0))[1];
    const float4 c10 = ((const float4*)(rowp + k0 + 16))[0];
    const float4 c11 = ((const float4*)(rowp + k0 + 16))[1];
    a[0]=(bf16_t)c00.x; a[1]=(bf16_t)c00.y; a[2]=(bf16_t)c00.z; a[3]=(bf16_t)c00.w;
    a[4]=(bf16_t)c01.x; a[5]=(bf16_t)c01.y; a[6]=(bf16_t)c01.z; a[7]=(bf16_t)c01.w;
    a[8]=(bf16_t)c10.x; a[9]=(bf16_t)c10.y; a[10]=(bf16_t)c10.z; a[11]=(bf16_t)c10.w;
    a[12]=(bf16_t)c11.x; a[13]=(bf16_t)c11.y; a[14]=(bf16_t)c11.z; a[15]=(bf16_t)c11.w;
  } else {
    #pragma unroll
    for (int i = 0; i < 16; ++i) a[i] = (bf16_t)0.0f;
  }
  return a;
}

// Same, but source is the agg buffer where -inf marks empty segments -> 0.
__device__ __forceinline__ v16bf make_a_agg(const float* __restrict__ rowp,
                                            int k0, bool ok) {
  v16bf a;
  if (ok) {
    #pragma unroll
    for (int i = 0; i < 8; ++i) {
      float v0 = rowp[k0 + i];
      float v1 = rowp[k0 + 16 + i];
      if (__float_as_uint(v0) == 0xFF800000u) v0 = 0.0f;
      if (__float_as_uint(v1) == 0xFF800000u) v1 = 0.0f;
      a[i]     = (bf16_t)v0;
      a[8 + i] = (bf16_t)v1;
    }
  } else {
    #pragma unroll
    for (int i = 0; i < 16; ++i) a[i] = (bf16_t)0.0f;
  }
  return a;
}

// A-fragment from staged bf16 hidden tile in LDS (row-major 16x128 per tile).
__device__ __forceinline__ v16bf make_a_lds(const bf16_t* __restrict__ hrow, int k0) {
  v16bf a;
  #pragma unroll
  for (int i = 0; i < 8; ++i) { a[i] = hrow[k0 + i]; a[8 + i] = hrow[k0 + 16 + i]; }
  return a;
}

// B fragment read from LDS (32 contiguous bytes per lane -> 2x ds_load_b128)
__device__ __forceinline__ v16bf load_b(const bf16_t* __restrict__ sW,
                                        int kt, int nt, int lane) {
  return *(const v16bf*)(sW + ((kt * NT + nt) * 32 + lane) * 16);
}

// Double-buffered B-fragment GEMM step over all 8 n-tiles:
// issue nt+1's LDS loads into the alternate buffer before consuming nt,
// so two ds_load_b128 pairs stay in flight (no full dscnt==0 serialization).
__device__ __forceinline__ void gemm_step(const bf16_t* __restrict__ sW, int kt,
                                          int lane, const v16bf a[ROWS],
                                          v8f acc[ROWS][NT]) {
  v16bf bcur = load_b(sW, kt, 0, lane);
  #pragma unroll
  for (int nt = 0; nt < NT; ++nt) {
    const v16bf bnext = (nt + 1 < NT) ? load_b(sW, kt, nt + 1, lane) : bcur;
    #pragma unroll
    for (int t = 0; t < ROWS; ++t)
      acc[t][nt] = __builtin_amdgcn_wmma_f32_16x16x32_bf16(
          false, a[t], false, bcur, (short)0, acc[t][nt], false, false);
    bcur = bnext;
  }
}

// ---------------------------------------------------------------------------
// Edge kernel: msg = ReLU(X @ We1 + be1) @ We2 + be2 ; scatter-max into agg.
// One wave per 32-edge pair of tiles; each LDS B fragment feeds 2 WMMAs.
// Dynamic LDS: We1 96KB + We2 32KB + per-wave 2-tile staging 64KB = 192KB.
// ---------------------------------------------------------------------------
__global__ void __launch_bounds__(256) edge_mlp_scatter(
    const float* __restrict__ X, const int* __restrict__ nodeIdx,
    const float* __restrict__ be1, const float* __restrict__ be2,
    const bf16_t* __restrict__ We1s, const bf16_t* __restrict__ We2s,
    float* __restrict__ agg, int E) {
  extern __shared__ __align__(32) char smem[];
  bf16_t* sW1   = (bf16_t*)smem;                             // WE1_ELEMS
  bf16_t* sW2   = (bf16_t*)(smem + (size_t)WE1_ELEMS * 2);   // WE2_ELEMS
  bf16_t* sHidA = (bf16_t*)(smem + (size_t)(WE1_ELEMS + WE2_ELEMS) * 2);

  // cooperative weight load (16B chunks)
  {
    const uint4* g1 = (const uint4*)We1s; uint4* l1 = (uint4*)sW1;
    for (int i = threadIdx.x; i < WE1_ELEMS / 8; i += blockDim.x) l1[i] = g1[i];
    const uint4* g2 = (const uint4*)We2s; uint4* l2 = (uint4*)sW2;
    for (int i = threadIdx.x; i < WE2_ELEMS / 8; i += blockDim.x) l2[i] = g2[i];
  }
  __syncthreads();

  const int wave = threadIdx.x >> 5;
  const int lane = threadIdx.x & 31;
  const int m    = lane & 15;
  const int hi   = lane >> 4;
  bf16_t* sHid   = sHidA + wave * (ROWS * 16 * Hh);

  const int pairs  = (E + 31) >> 5;
  const int stride = gridDim.x * 8;

  for (int p = blockIdx.x * 8 + wave; p < pairs; p += stride) {
    const int base = p * 32;
    if (p + stride < pairs)
      __builtin_prefetch(X + (size_t)(p + stride) * 32 * K1 + lane * 16, 0, 1);

    // ---- GEMM1: [32 x 384] @ [384 x 128] ----
    v8f acc[ROWS][NT] = {};
    for (int kt = 0; kt < KT1; ++kt) {
      const int k0 = kt * 32 + hi * 8;
      v16bf a[ROWS];
      #pragma unroll
      for (int t = 0; t < ROWS; ++t) {
        const int row = base + t * 16 + m;
        a[t] = make_a_f32(X + (size_t)row * K1, k0, row < E);
      }
      gemm_step(sW1, kt, lane, a, acc);
    }

    // ---- bias + ReLU, stage hidden tiles (C layout -> row-major bf16) ----
    #pragma unroll
    for (int t = 0; t < ROWS; ++t) {
      #pragma unroll
      for (int nt = 0; nt < NT; ++nt) {
        const int col = nt * 16 + m;
        const float b1 = be1[col];
        #pragma unroll
        for (int j = 0; j < 8; ++j) {
          float v = acc[t][nt][j] + b1;       // C layout: M = j + 8*hi, N = col
          v = v > 0.0f ? v : 0.0f;
          sHid[(t * 16 + j + hi * 8) * Hh + col] = (bf16_t)v;
        }
      }
    }
    // per-wave staging; LDS ops from the same wave are in-order (DScnt)

    // ---- GEMM2: [32 x 128] @ [128 x 128] ----
    v8f acc2[ROWS][NT] = {};
    #pragma unroll
    for (int kt = 0; kt < KTH; ++kt) {
      const int k0 = kt * 32 + hi * 8;
      v16bf a[ROWS];
      #pragma unroll
      for (int t = 0; t < ROWS; ++t)
        a[t] = make_a_lds(sHid + (t * 16 + m) * Hh, k0);
      gemm_step(sW2, kt, lane, a, acc2);
    }

    // ---- bias + scatter-max into agg ----
    #pragma unroll
    for (int t = 0; t < ROWS; ++t) {
      int nidx[8];
      #pragma unroll
      for (int j = 0; j < 8; ++j) {
        const int r = base + t * 16 + j + hi * 8;
        nidx[j] = (r < E) ? nodeIdx[r] : -1;
      }
      #pragma unroll
      for (int nt = 0; nt < NT; ++nt) {
        const int col = nt * 16 + m;
        const float b2 = be2[col];
        #pragma unroll
        for (int j = 0; j < 8; ++j) {
          if (nidx[j] >= 0)
            atomic_fmax(agg + (size_t)nidx[j] * Hh + col, acc2[t][nt][j] + b2);
        }
      }
    }
  }
}

// ---------------------------------------------------------------------------
// Node kernel: lang = ReLU([center|agg] @ Wf1 + bf1) @ Wf2 + bf2 ;
// out[r] = cos(gcn[r], lang[r]) with eps clamp. One wave per 32-node pair.
// Dynamic LDS: Wf1 64KB + Wf2 32KB + 2-tile staging 64KB = 160KB.
// ---------------------------------------------------------------------------
__global__ void __launch_bounds__(256) node_update_cos(
    const float* __restrict__ center, const float* __restrict__ gcn,
    const float* __restrict__ agg,
    const float* __restrict__ bf1, const float* __restrict__ bf2,
    const bf16_t* __restrict__ Wf1s, const bf16_t* __restrict__ Wf2s,
    float* __restrict__ out, int N) {
  extern __shared__ __align__(32) char smem[];
  bf16_t* sW1   = (bf16_t*)smem;                             // WF1_ELEMS
  bf16_t* sW2   = (bf16_t*)(smem + (size_t)WF1_ELEMS * 2);   // WF2_ELEMS
  bf16_t* sHidA = (bf16_t*)(smem + (size_t)(WF1_ELEMS + WF2_ELEMS) * 2);

  {
    const uint4* g1 = (const uint4*)Wf1s; uint4* l1 = (uint4*)sW1;
    for (int i = threadIdx.x; i < WF1_ELEMS / 8; i += blockDim.x) l1[i] = g1[i];
    const uint4* g2 = (const uint4*)Wf2s; uint4* l2 = (uint4*)sW2;
    for (int i = threadIdx.x; i < WF2_ELEMS / 8; i += blockDim.x) l2[i] = g2[i];
  }
  __syncthreads();

  const int wave = threadIdx.x >> 5;
  const int lane = threadIdx.x & 31;
  const int m    = lane & 15;
  const int hi   = lane >> 4;
  bf16_t* sHid   = sHidA + wave * (ROWS * 16 * Hh);

  const int pairs  = (N + 31) >> 5;
  const int stride = gridDim.x * 8;

  for (int p = blockIdx.x * 8 + wave; p < pairs; p += stride) {
    const int base = p * 32;

    // ---- GEMM1: [32 x 256] @ [256 x 128] ; K<128 from center, else agg ----
    v8f acc[ROWS][NT] = {};
    for (int kt = 0; kt < KT2; ++kt) {
      const int k0 = kt * 32 + hi * 8;
      v16bf a[ROWS];
      #pragma unroll
      for (int t = 0; t < ROWS; ++t) {
        const int row = base + t * 16 + m;
        const bool ok = row < N;
        if (kt < KT2 / 2)
          a[t] = make_a_f32(center + (size_t)row * Hh, k0, ok);
        else
          a[t] = make_a_agg(agg + (size_t)row * Hh, k0 - Hh, ok);
      }
      gemm_step(sW1, kt, lane, a, acc);
    }

    // ---- bias + ReLU, stage hidden ----
    #pragma unroll
    for (int t = 0; t < ROWS; ++t) {
      #pragma unroll
      for (int nt = 0; nt < NT; ++nt) {
        const int col = nt * 16 + m;
        const float b1 = bf1[col];
        #pragma unroll
        for (int j = 0; j < 8; ++j) {
          float v = acc[t][nt][j] + b1;
          v = v > 0.0f ? v : 0.0f;
          sHid[(t * 16 + j + hi * 8) * Hh + col] = (bf16_t)v;
        }
      }
    }

    // ---- GEMM2: [32 x 128] @ [128 x 128] ----
    v8f acc2[ROWS][NT] = {};
    #pragma unroll
    for (int kt = 0; kt < KTH; ++kt) {
      const int k0 = kt * 32 + hi * 8;
      v16bf a[ROWS];
      #pragma unroll
      for (int t = 0; t < ROWS; ++t)
        a[t] = make_a_lds(sHid + (t * 16 + m) * Hh, k0);
      gemm_step(sW2, kt, lane, a, acc2);
    }

    // ---- cosine similarity straight from C layout ----
    // lane holds lang(M = j + 8*hi, N = nt*16 + m); reduce over the 16-lane half
    #pragma unroll
    for (int t = 0; t < ROWS; ++t) {
      #pragma unroll
      for (int j = 0; j < 8; ++j) {
        const int r = base + t * 16 + j + hi * 8;  // same r across the 16 lanes
        float d = 0.0f, l2 = 0.0f, g2 = 0.0f;
        if (r < N) {
          #pragma unroll
          for (int nt = 0; nt < NT; ++nt) {
            const int col = nt * 16 + m;
            const float lv = acc2[t][nt][j] + bf2[col];
            const float gv = gcn[(size_t)r * Hh + col];
            d += lv * gv; l2 += lv * lv; g2 += gv * gv;
          }
        }
        #pragma unroll
        for (int s = 1; s < 16; s <<= 1) {    // xor stays inside the half-wave
          d  += __shfl_xor(d,  s, 32);
          l2 += __shfl_xor(l2, s, 32);
          g2 += __shfl_xor(g2, s, 32);
        }
        if (m == 0 && r < N) {
          const float na = fmaxf(__builtin_sqrtf(g2), 1e-8f);
          const float nb = fmaxf(__builtin_sqrtf(l2), 1e-8f);
          out[r] = d / (na * nb);
        }
      }
    }
  }
}

// ---------------------------------------------------------------------------
extern "C" void kernel_launch(void* const* d_in, const int* in_sizes, int n_in,
                              void* d_out, int out_size, void* d_ws, size_t ws_size,
                              hipStream_t stream) {
  const float* center = (const float*)d_in[0];
  const float* leaf   = (const float*)d_in[1];
  const int*   nidx   = (const int*)d_in[2];
  const float* gcn    = (const float*)d_in[3];
  const float* We1 = (const float*)d_in[4];  const float* be1 = (const float*)d_in[5];
  const float* We2 = (const float*)d_in[6];  const float* be2 = (const float*)d_in[7];
  const float* Wf1 = (const float*)d_in[8];  const float* bf1 = (const float*)d_in[9];
  const float* Wf2 = (const float*)d_in[10]; const float* bf2 = (const float*)d_in[11];
  float* out = (float*)d_out;

  const int N = in_sizes[0] / Hh;
  const int E = in_sizes[1] / K1;

  // workspace layout: agg[N*128] f32, then swizzled bf16 weights
  char* ws = (char*)d_ws;
  float* agg = (float*)ws;
  size_t off = (((size_t)N * Hh * sizeof(float)) + 255) & ~(size_t)255;
  bf16_t* We1s = (bf16_t*)(ws + off);
  bf16_t* We2s = We1s + WE1_ELEMS;
  bf16_t* Wf1s = We2s + WE2_ELEMS;
  bf16_t* Wf2s = Wf1s + WF1_ELEMS;

  const int prepElems = WE1_ELEMS + WE2_ELEMS + WF1_ELEMS + WF2_ELEMS;
  prep_weights<<<(prepElems + 255) / 256, 256, 0, stream>>>(
      We1, We2, Wf1, Wf2, We1s, We2s, Wf1s, Wf2s);

  const int aggElems = N * Hh;
  init_agg<<<(aggElems + 255) / 256, 256, 0, stream>>>((unsigned int*)agg, aggElems);

  const int pairsE = (E + 31) / 32;
  int blocksE = (pairsE + 7) / 8; if (blocksE > 640) blocksE = 640;
  const size_t ldsE =
      (size_t)(WE1_ELEMS + WE2_ELEMS) * 2 + 8 * ROWS * 16 * Hh * 2;  // 192 KB
  edge_mlp_scatter<<<blocksE, 256, ldsE, stream>>>(leaf, nidx, be1, be2, We1s, We2s, agg, E);

  const int pairsN = (N + 31) / 32;
  int blocksN = (pairsN + 7) / 8; if (blocksN > 512) blocksN = 512;
  const size_t ldsN =
      (size_t)(WF1_ELEMS + WF2_ELEMS) * 2 + 8 * ROWS * 16 * Hh * 2;  // 160 KB
  node_update_cos<<<blocksN, 256, ldsN, stream>>>(center, gcn, agg, bf1, bf2, Wf1s, Wf2s, out, N);
}
